// TreeTextDecoder_31550829756814
// MI455X (gfx1250) — matrix operand order repaired and checked by reading, not compiled, not run on hardware
//
#include <hip/hip_runtime.h>
#include <hip/hip_bf16.h>

typedef __attribute__((ext_vector_type(16))) _Float16 v16h;
typedef __attribute__((ext_vector_type(8)))  _Float16 h8;
typedef __attribute__((ext_vector_type(8)))  float    v8f;

#define D     256
#define BATCH 8192
#define SEQ   64
#define MT    16          // batch rows per workgroup
#define RS    264         // padded row stride in halfs (16B pad -> conflict-free ds_load_b128)
#define EPS   1e-5f

// ---------------- prep: convert weights to f16, precompute spike scale, zero stats ----------------
__global__ __launch_bounds__(256) void prep_kernel(const float* __restrict__ W_nodes,
                                                   const float* __restrict__ W_out,
                                                   const float* __restrict__ thr,
                                                   const float* __restrict__ vre,
                                                   _Float16* __restrict__ Wh,
                                                   _Float16* __restrict__ Whout,
                                                   float* __restrict__ scalebuf,
                                                   float* __restrict__ stats) {
  int i = blockIdx.x * 256 + threadIdx.x;
  if (i < 15 * D * D) Wh[i] = (_Float16)W_nodes[i];
  if (i < SEQ * D)    Whout[i] = (_Float16)W_out[i];
  if (i < 15 * D)     scalebuf[i] = 1.0f / (1.0f + __expf(-25.0f * (vre[i] - thr[i])));
  if (i < 2)          stats[i] = 0.0f;
}

// ---------------- main: full tree + LN + output matmul for one 16-row batch tile ----------------
__global__ __launch_bounds__(256) void tree_kernel(const float* __restrict__ x,
                                                   const float* __restrict__ b_nodes,
                                                   const float* __restrict__ thr,
                                                   const float* __restrict__ ln_gamma,
                                                   const float* __restrict__ ln_beta,
                                                   const float* __restrict__ b_out,
                                                   const _Float16* __restrict__ Wh,
                                                   const _Float16* __restrict__ Whout,
                                                   const float* __restrict__ scalebuf,
                                                   float* __restrict__ y_ws,
                                                   float* __restrict__ stats) {
  __shared__ __align__(16) _Float16 xin[MT][RS];
  __shared__ __align__(16) _Float16 bufA[8][MT][RS];
  __shared__ __align__(16) _Float16 bufB[4][MT][RS];
  __shared__ __align__(16) float    rootf[MT][RS];
  __shared__ __align__(16) _Float16 rootln[MT][RS];

  const int tid  = threadIdx.x;
  const int wave = tid >> 5;
  const int lane = tid & 31;
  const int ln   = lane & 15;   // row (A) / column (B,C) index within 16x16 tile
  const int hi   = lane >> 4;   // lane half selects K-half / M-half
  const int m0   = blockIdx.x * MT;

  // stage x tile into LDS as f16
  for (int i = tid; i < MT * D; i += 256) {
    int r = i >> 8, c = i & (D - 1);
    xin[r][c] = (_Float16)x[(size_t)(m0 + r) * D + c];
  }
  __syncthreads();

  // spike-activation epilogue for one 16x16 C tile
  auto epilogue = [&](v8f acc, int widx, int n, _Float16* Out, bool toRoot) {
    const float bias = b_nodes[widx * D + n];
    const float tv   = thr[widx * D + n];
    const float sv   = scalebuf[widx * D + n];
    #pragma unroll
    for (int g = 0; g < 8; ++g) {
      float z = acc[g] + bias;
      float o = (z >= tv) ? z * sv : 0.0f;
      int row = g + 8 * hi;               // C layout: VGPR g -> M = g + 8*hi, N = ln
      if (toRoot) rootf[row][n] = o;
      else        Out[row * RS + n] = (_Float16)o;
    }
  };

  // one node: (16 x 256) @ W[widx]^T + b, spike activation, store f16 (or f32 root).
  // Each wave computes two 16-wide N-tiles sharing the A-operand loads.
  auto node_mm = [&](const _Float16* In, int widx, int widx_next, _Float16* Out, bool toRoot) {
    const _Float16* Wn = Wh + (size_t)widx * D * D;
    const int n0 = 16 * (2 * wave) + ln;
    const int n1 = n0 + 16;
    const _Float16* wp0 = Wn + (size_t)n0 * D + 16 * hi;
    const _Float16* wp1 = Wn + (size_t)n1 * D + 16 * hi;
    // prefetch next node's weight rows (real latency distance, near scope)
    if (widx_next >= 0) {
      const _Float16* Wnx = Wh + (size_t)widx_next * D * D;
      __builtin_prefetch(Wnx + (size_t)n0 * D, 0, 3);
      __builtin_prefetch(Wnx + (size_t)n1 * D, 0, 3);
    }
    v8f acc0 = {}, acc1 = {};
    #pragma unroll
    for (int kb = 0; kb < D; kb += 32) {
      v16h a, b0, b1;
      // A: 16x32 f16 tile, row M=ln; VGPR0-7 = K(kb+16*hi ..+15)
      *(h8*)&a          = *(const h8*)(In + ln * RS + kb + 16 * hi);
      *(((h8*)&a) + 1)  = *(const h8*)(In + ln * RS + kb + 16 * hi + 8);
      // B: 32x16, column N=n; 16 contiguous halfs of W row n starting at kb+16*hi
      *(h8*)&b0         = *(const h8*)(wp0 + kb);
      *(((h8*)&b0) + 1) = *(const h8*)(wp0 + kb + 8);
      *(h8*)&b1         = *(const h8*)(wp1 + kb);
      *(((h8*)&b1) + 1) = *(const h8*)(wp1 + kb + 8);
      acc0 = __builtin_amdgcn_wmma_f32_16x16x32_f16(false, a, false, b0,
                                                    (short)0, acc0, false, false);
      acc1 = __builtin_amdgcn_wmma_f32_16x16x32_f16(false, a, false, b1,
                                                    (short)0, acc1, false, false);
    }
    epilogue(acc0, widx, n0, Out, toRoot);
    epilogue(acc1, widx, n1, Out, toRoot);
  };

  // pairwise average: nout fused nodes from 2*nout source nodes
  auto fuse = [&](const _Float16* src, _Float16* dst, int nout) {
    const int chunks = D / 8;
    const int total = nout * MT * chunks;
    for (int i = tid; i < total; i += 256) {
      int j   = i / (MT * chunks);
      int rem = i - j * MT * chunks;
      int r   = rem / chunks;
      int c8  = (rem - r * chunks) * 8;
      const _Float16* pa = src + ((2 * j) * MT + r) * RS + c8;
      const _Float16* pb = src + ((2 * j + 1) * MT + r) * RS + c8;
      _Float16* pd = dst + (j * MT + r) * RS + c8;
      #pragma unroll
      for (int e = 0; e < 8; ++e)
        pd[e] = (_Float16)(((float)pa[e] + (float)pb[e]) * 0.5f);
    }
  };

  // leaf level: 8 nodes, W[7..14], shared input x
  for (int nd = 0; nd < 8; ++nd)
    node_mm(&xin[0][0], 7 + nd, (nd < 7) ? 8 + nd : 3, &bufA[nd][0][0], false);
  __syncthreads();
  fuse(&bufA[0][0][0], &bufB[0][0][0], 4);
  __syncthreads();
  for (int nd = 0; nd < 4; ++nd)
    node_mm(&bufB[nd][0][0], 3 + nd, (nd < 3) ? 4 + nd : 1, &bufA[nd][0][0], false);
  __syncthreads();
  fuse(&bufA[0][0][0], &bufB[0][0][0], 2);
  __syncthreads();
  for (int nd = 0; nd < 2; ++nd)
    node_mm(&bufB[nd][0][0], 1 + nd, (nd < 1) ? 2 : 0, &bufA[nd][0][0], false);
  __syncthreads();
  fuse(&bufA[0][0][0], &bufB[0][0][0], 1);
  __syncthreads();
  node_mm(&bufB[0][0][0], 0, -1, nullptr, true);   // root -> rootf (f32)
  __syncthreads();

  // LayerNorm per row: wave handles rows 2*wave (+hi), 16 lanes per row
  {
    int row = 2 * wave + hi;
    float s = 0.0f, sq = 0.0f;
    for (int c = ln; c < D; c += 16) { float v = rootf[row][c]; s += v; sq += v * v; }
    #pragma unroll
    for (int off = 1; off < 16; off <<= 1) {
      s  += __shfl_xor(s, off, 32);
      sq += __shfl_xor(sq, off, 32);
    }
    float mu  = s * (1.0f / D);
    float var = sq * (1.0f / D) - mu * mu;
    float inv = rsqrtf(var + EPS);
    for (int c = ln; c < D; c += 16) {
      float v = (rootf[row][c] - mu) * inv * ln_gamma[c] + ln_beta[c];
      rootln[row][c] = (_Float16)v;
    }
  }
  __syncthreads();

  // output matmul: y = rootln @ W_out^T + b_out  (N=64 -> waves 0..3, one 16-col tile each)
  if (wave < 4) {
    const int nb = 16 * wave;
    const int n  = nb + ln;                 // seq index 0..63
    const _Float16* wp = Whout + (size_t)n * D + 16 * hi;
    v8f acc = {};
    #pragma unroll
    for (int kb = 0; kb < D; kb += 32) {
      v16h a, b;
      *(h8*)&a         = *(const h8*)(&rootln[0][0] + ln * RS + kb + 16 * hi);
      *(((h8*)&a) + 1) = *(const h8*)(&rootln[0][0] + ln * RS + kb + 16 * hi + 8);
      *(h8*)&b         = *(const h8*)(wp + kb);
      *(((h8*)&b) + 1) = *(const h8*)(wp + kb + 8);
      acc = __builtin_amdgcn_wmma_f32_16x16x32_f16(false, a, false, b,
                                                   (short)0, acc, false, false);
    }
    const float bias = b_out[n];
    float ls = 0.0f, lsq = 0.0f;
    #pragma unroll
    for (int g = 0; g < 8; ++g) {
      float yv = acc[g] + bias;
      int row = g + 8 * hi;
      y_ws[(size_t)(m0 + row) * SEQ + n] = yv;
      ls += yv; lsq += yv * yv;
    }
    #pragma unroll
    for (int off = 1; off < 32; off <<= 1) {
      ls  += __shfl_xor(ls, off, 32);
      lsq += __shfl_xor(lsq, off, 32);
    }
    if (lane == 0) {
      atomicAdd(&stats[0], ls);
      atomicAdd(&stats[1], lsq);
    }
  }
}

// ---------------- finalize: global mean/var normalization of y ----------------
__global__ __launch_bounds__(256) void finalize_kernel(const float* __restrict__ y_ws,
                                                       const float* __restrict__ stats,
                                                       float* __restrict__ out) {
  const int i = blockIdx.x * 256 + threadIdx.x;
  const float N = (float)(BATCH * SEQ);
  float m = stats[0] / N;
  float v = stats[1] / N - m * m;
  float inv = rsqrtf(v + EPS);
  float yv = y_ws[i];
  out[i] = (yv - m) * inv + yv;
}

extern "C" void kernel_launch(void* const* d_in, const int* in_sizes, int n_in,
                              void* d_out, int out_size, void* d_ws, size_t ws_size,
                              hipStream_t stream) {
  const float* fused_x  = (const float*)d_in[0];
  const float* W_nodes  = (const float*)d_in[1];
  const float* b_nodes  = (const float*)d_in[2];
  const float* threshold= (const float*)d_in[3];
  // d_in[4] = tau (unused by reference forward)
  const float* v_reset  = (const float*)d_in[5];
  const float* ln_gamma = (const float*)d_in[6];
  const float* ln_beta  = (const float*)d_in[7];
  const float* W_out    = (const float*)d_in[8];
  const float* b_out    = (const float*)d_in[9];

  char* ws = (char*)d_ws;
  size_t off = 0;
  float* stats = (float*)(ws + off);              off += 16;
  float* scalebuf = (float*)(ws + off);           off += (size_t)15 * D * sizeof(float);
  _Float16* Wh = (_Float16*)(ws + off);           off += (size_t)15 * D * D * sizeof(_Float16);
  _Float16* Whout = (_Float16*)(ws + off);        off += (size_t)SEQ * D * sizeof(_Float16);
  float* y_ws = (float*)(ws + off);               off += (size_t)BATCH * SEQ * sizeof(float);

  // prep: 15*256*256 = 983040 elements -> 3840 blocks
  prep_kernel<<<3840, 256, 0, stream>>>(W_nodes, W_out, threshold, v_reset,
                                        Wh, Whout, scalebuf, stats);

  tree_kernel<<<BATCH / MT, 256, 0, stream>>>(fused_x, b_nodes, threshold,
                                              ln_gamma, ln_beta, b_out,
                                              Wh, Whout, scalebuf, y_ws, stats);

  finalize_kernel<<<(BATCH * SEQ) / 256, 256, 0, stream>>>(y_ws, stats, (float*)d_out);
}
